// WrapSegmentationNet_30880814858574
// MI455X (gfx1250) — matrix operand order repaired and checked by reading, not compiled, not run on hardware
//
#include <hip/hip_runtime.h>
#include <hip/hip_bf16.h>
#include <math.h>

typedef _Float16 f16;
typedef __attribute__((ext_vector_type(16))) _Float16 v16h;
typedef __attribute__((ext_vector_type(8)))  _Float16 v8h;
typedef __attribute__((ext_vector_type(8)))  float    v8f;

// ---------------------------------------------------------------------------
// WMMA helpers (CDNA5 v_wmma_f32_16x16x32_f16, wave32)
//  A (16x32 f16): lane<16 -> M=lane, K={kb..kb+7, kb+16..kb+23};
//                 lane>=16 -> M=lane-16, K={kb+8..kb+15, kb+24..kb+31}
//  B (32x16 f16): lane<16 -> N=lane, K={kb..kb+7, kb+16..kb+23}; lane>=16 -> K={kb+8..kb+15, kb+24..kb+31}
//  C (16x16 f32): elem q: M = ((lane>>4)&1)*8 + q, N = lane&15
// All fragment data staged in LDS as f16 -> aligned b128 reads, no masks.
// ---------------------------------------------------------------------------

static __device__ inline v16h pack16(v8h lo, v8h hi) {
  v16h a;
#pragma unroll
  for (int j = 0; j < 8; ++j) { a[j] = lo[j]; a[j + 8] = hi[j]; }
  return a;
}

static __device__ inline v16h frag_a_lds(const f16* row, int kb, int lane) {
  int koff = (lane & 16) ? 8 : 0;
  v8h lo = *(const v8h*)(row + kb + koff);
  v8h hi = *(const v8h*)(row + kb + 16 + koff);
  return pack16(lo, hi);
}

static __device__ inline v16h frag_b_lds(const f16* col, int kb, int lane) {
  const f16* p = col + kb + ((lane & 16) ? 8 : 0);
  v8h lo = *(const v8h*)(p);
  v8h hi = *(const v8h*)(p + 16);
  return pack16(lo, hi);
}

static __device__ inline v8f wmma_f16(v16h a, v16h b, v8f c) {
  return __builtin_amdgcn_wmma_f32_16x16x32_f16(
      false, a, false, b, (short)0, c, false, false);
}

// ---------------------------------------------------------------------------
#define BATCH 32
#define NPTS  1024
#define NODES (BATCH * NPTS)
#define KNN   30

// ---------------------------------------------------------------------------
__global__ void sq_kernel(const float* __restrict__ x, float* __restrict__ sq,
                          int total, int d) {
  int t = blockIdx.x * blockDim.x + threadIdx.x;
  if (t >= total) return;
  float s = 0.f;
  for (int k = 0; k < d; ++k) {
    float v = x[(size_t)t * d + k];
    s += v * v;
  }
  sq[t] = s;
}

// ---------------------------------------------------------------------------
// kNN: 128 thr (4 waves) per (batch, 16-row strip).  Staging fully
// specialized per D so no runtime masks exist anywhere.
// ---------------------------------------------------------------------------
template <int D>
__global__ void __launch_bounds__(128)
knn_kernel(const float* __restrict__ x, const float* __restrict__ sq,
           int* __restrict__ idx, int N) {
  constexpr int DP  = (D <= 32) ? 32 : 64;
  constexpr int STR = DP + 8;
  constexpr int KS  = DP / 32;

  extern __shared__ char smem[];
  float* ds2 = (float*)smem;                         // [16][NPTS]
  f16*   sA  = (f16*)(smem + 16 * NPTS * 4);         // [16][STR]
  f16*   sB  = sA + 16 * STR;                        // [4][16][STR]

  const int ntiles_i = N >> 4;
  int b  = blockIdx.x / ntiles_i;
  int i0 = (blockIdx.x % ntiles_i) * 16;
  const float* xb  = x  + (size_t)b * N * D;
  const float* sqb = sq + (size_t)b * N;
  int lane = threadIdx.x & 31;
  int wave = threadIdx.x >> 5;

  // ---- stage A rows (f16, zero-padded), compile-time masks only ----
  if (D == 64) {
    int row = threadIdx.x >> 3;
    int c0  = (threadIdx.x & 7) * 8;
    const float* src = xb + (size_t)(i0 + row) * D + c0;
    v8h h;
#pragma unroll
    for (int j = 0; j < 8; ++j) h[j] = (f16)src[j];
    *(v8h*)(sA + row * STR + c0) = h;
  } else if (threadIdx.x < 16) {     // D == 7, DP == 32
    int row = threadIdx.x;
    const float* src = xb + (size_t)(i0 + row) * D;
#pragma unroll
    for (int c = 0; c < DP; ++c)
      sA[row * STR + c] = (f16)((c < D) ? src[c] : 0.f);
  }
  __syncthreads();

  f16* sBw = sB + wave * 16 * STR;
  for (int jt = wave; jt < (N >> 4); jt += 4) {
    int j0 = jt * 16;
    // ---- stage this wave's B tile: sBw[n][k] = X[j0+n][k] ----
    if (D == 64) {
      int r  = lane >> 1;
      int c0 = (lane & 1) * 32;
      const float* src = xb + (size_t)(j0 + r) * D + c0;
#pragma unroll
      for (int cc = 0; cc < 32; cc += 8) {
        v8h h;
#pragma unroll
        for (int j = 0; j < 8; ++j) h[j] = (f16)src[cc + j];
        *(v8h*)(sBw + r * STR + c0 + cc) = h;
      }
    } else if (lane < 16) {          // D == 7
      int r = lane;
      const float* src = xb + (size_t)(j0 + r) * D;
#pragma unroll
      for (int c = 0; c < DP; ++c)
        sBw[r * STR + c] = (f16)((c < D) ? src[c] : 0.f);
    }
    // per-wave LDS store->load is in-order; no barrier needed
    v8f acc = {};
#pragma unroll
    for (int ks = 0; ks < KS; ++ks) {
      v16h a  = frag_a_lds(sA + (lane & 15) * STR, ks * 32, lane);
      v16h bf = frag_b_lds(sBw + (lane & 15) * STR, ks * 32, lane);
      acc = wmma_f16(a, bf, acc);
    }
    int n = lane & 15;
    float sqj = sqb[j0 + n];
#pragma unroll
    for (int q = 0; q < 8; ++q) {
      int M = ((lane >> 4) & 1) * 8 + q;
      ds2[M * NPTS + j0 + n] = sqb[i0 + M] + sqj - 2.0f * acc[q];
    }
  }
  __syncthreads();

  if (threadIdx.x < 16) {
    int r = threadIdx.x;
    const float* drow = ds2 + r * NPTS;
    float bv[KNN];
    int   bi[KNN];
    for (int q = 0; q < KNN; ++q) { bv[q] = 3.0e38f; bi[q] = 0; }
    for (int j = 0; j < N; ++j) {
      float v = drow[j];
      if (v < bv[KNN - 1]) {
        int p = KNN - 1;
        while (p > 0 && bv[p - 1] > v) {
          bv[p] = bv[p - 1]; bi[p] = bi[p - 1]; --p;
        }
        bv[p] = v; bi[p] = j;
      }
    }
    int* op = idx + ((size_t)b * N + i0 + r) * KNN;
    for (int q = 0; q < KNN; ++q) op[q] = bi[q];
  }
}

// ---------------------------------------------------------------------------
// Fused DynamicEdgeConv: 8 waves/block, one node per wave.
// ---------------------------------------------------------------------------
template <int D>
__global__ void __launch_bounds__(256)
edgeconv_kernel(const float* __restrict__ x, const int* __restrict__ idx,
                const float* __restrict__ W1, const float* __restrict__ b1,
                const float* __restrict__ sc, const float* __restrict__ sh,
                const float* __restrict__ W2, const float* __restrict__ b2,
                float* __restrict__ xout, int N) {
  constexpr int D2   = 2 * D;
  constexpr int DP   = (D2 <= 32) ? 32 : 128;
  constexpr int ASTR = DP + 8;
  constexpr int HSTR = 72;
  constexpr int KS1  = DP / 32;

  extern __shared__ char smem[];
  f16* sW1 = (f16*)smem;                 // [64][ASTR] col-major (n-major)
  f16* sW2 = sW1 + 64 * ASTR;            // [64][HSTR] col-major
  f16* sAa = sW2 + 64 * HSTR;            // 8 x [32][ASTR]
  f16* sHa = sAa + 8 * 32 * ASTR;        // 8 x [32][HSTR]

  int tid  = threadIdx.x;
  int wave = tid >> 5;
  int lane = tid & 31;

  // ---- stage W1 col-major, K zero-padded to DP ----
  for (int e = tid; e < DP * 64; e += 256) {
    int k = e >> 6, n = e & 63;
    float v = (k < D2) ? W1[(size_t)k * 64 + n] : 0.f;
    sW1[n * ASTR + k] = (f16)v;
  }
  // ---- stage W2 col-major ----
  for (int e = tid; e < 64 * 64; e += 256) {
    int k = e >> 6, n = e & 63;
    sW2[n * HSTR + k] = (f16)W2[e];
  }

  // ---- stage this wave's edge-feature tile ----
  int node = blockIdx.x * 8 + wave;
  int b = node / N, i = node % N;
  const float* xg = x + (size_t)b * N * D;
  const int*   ip = idx + (size_t)node * KNN;
  f16* sA = sAa + wave * 32 * ASTR;
  f16* sH = sHa + wave * 32 * HSTR;

  if (D == 64) {
    float xi0 = xg[(size_t)i * D + lane];
    float xi1 = xg[(size_t)i * D + 32 + lane];
    for (int e = 0; e < KNN; ++e) {
      const float* xj = xg + (size_t)ip[e] * D;
      f16* row = sA + e * ASTR;
      row[lane]      = (f16)xi0;
      row[32 + lane] = (f16)xi1;
      row[64 + lane] = (f16)(xj[lane] - xi0);
      row[96 + lane] = (f16)(xj[32 + lane] - xi1);
    }
    for (int c = lane; c < DP; c += 32) {   // pad rows 30,31
      sA[30 * ASTR + c] = (f16)0.f;
      sA[31 * ASTR + c] = (f16)0.f;
    }
  } else {  // D == 7: one lane per edge row; single predicate region
    float xi[7];
#pragma unroll
    for (int c = 0; c < 7; ++c) xi[c] = xg[(size_t)i * 7 + c];
    int e = lane;                          // rows 30,31 become zero padding
    bool real = (e < KNN);
    int ee = real ? e : 0;
    const float* xj = xg + (size_t)ip[ee] * 7;
    f16* row = sA + e * ASTR;
#pragma unroll
    for (int c = 0; c < 7; ++c) {
      float xjc = real ? xj[c] : 0.f;
      row[c]     = (f16)(real ? xi[c] : 0.f);
      row[7 + c] = (f16)(real ? (xjc - xi[c]) : 0.f);
    }
#pragma unroll
    for (int c = 14; c < 32; ++c) row[c] = (f16)0.f;
  }
  __syncthreads();

  // ---- GEMM1: edges @ W1, fused BN + ReLU -> sH (f16) ----
  for (int nt = 0; nt < 4; ++nt) {
    for (int mt = 0; mt < 2; ++mt) {
      v8f acc = {};
#pragma unroll
      for (int ks = 0; ks < KS1; ++ks) {
        v16h a  = frag_a_lds(sA + (mt * 16 + (lane & 15)) * ASTR, ks * 32, lane);
        v16h bf = frag_b_lds(sW1 + (nt * 16 + (lane & 15)) * ASTR, ks * 32, lane);
        acc = wmma_f16(a, bf, acc);
      }
      int n = nt * 16 + (lane & 15);
      float bb = b1[n], ss = sc[n], tt = sh[n];
#pragma unroll
      for (int q = 0; q < 8; ++q) {
        int M = ((lane >> 4) & 1) * 8 + q;
        float h = (acc[q] + bb) * ss + tt;
        sH[(mt * 16 + M) * HSTR + n] = (f16)fmaxf(h, 0.f);
      }
    }
  }
  // per-wave LDS traffic is in-order; no block barrier needed here

  // ---- GEMM2: H @ W2 + b2, masked max over 30 edges ----
  for (int nt = 0; nt < 4; ++nt) {
    float colmax = -3.0e38f;
    for (int mt = 0; mt < 2; ++mt) {
      v8f acc = {};
#pragma unroll
      for (int ks = 0; ks < 2; ++ks) {
        v16h a  = frag_a_lds(sH + (mt * 16 + (lane & 15)) * HSTR, ks * 32, lane);
        v16h bf = frag_b_lds(sW2 + (nt * 16 + (lane & 15)) * HSTR, ks * 32, lane);
        acc = wmma_f16(a, bf, acc);
      }
      int n = nt * 16 + (lane & 15);
      float bb = b2[n];
#pragma unroll
      for (int q = 0; q < 8; ++q) {
        int M = ((lane >> 4) & 1) * 8 + q;
        if (mt * 16 + M < KNN) colmax = fmaxf(colmax, acc[q] + bb);
      }
    }
    colmax = fmaxf(colmax, __shfl_xor(colmax, 16, 32));
    if (lane < 16) xout[(size_t)node * 64 + nt * 16 + lane] = colmax;
  }
}

// ---------------------------------------------------------------------------
// Head GEMM: 64x64 macro-tile per 128-thr block, K staged in 32-chunks.
// ---------------------------------------------------------------------------
template <typename AT>
static __device__ inline void stage_row16(const AT* src, f16* dst) {
  v8h a, b;
#pragma unroll
  for (int j = 0; j < 8; ++j) a[j] = (f16)(float)src[j];
#pragma unroll
  for (int j = 0; j < 8; ++j) b[j] = (f16)(float)src[8 + j];
  *(v8h*)dst = a;
  *(v8h*)(dst + 8) = b;
}

template <typename AT, typename OT>
__global__ void __launch_bounds__(128)
gemm_kernel(const AT* __restrict__ A, const float* __restrict__ W,
            const float* __restrict__ bias, OT* __restrict__ out,
            int M, int K, int N, int relu, int nblk64) {
  __shared__ __align__(16) f16 sA[64 * 40];
  __shared__ __align__(16) f16 sB[64 * 40];
  int mblk = blockIdx.x / nblk64;
  int nblk = blockIdx.x % nblk64;
  int m0 = mblk * 64, n0 = nblk * 64;
  int tid = threadIdx.x, wave = tid >> 5, lane = tid & 31;

  v8f accs[4] = {};
  for (int kb = 0; kb < K; kb += 32) {
    {
      int row = tid >> 1, c0 = (tid & 1) * 16;
      stage_row16(A + (size_t)(m0 + row) * K + kb + c0, sA + row * 40 + c0);
    }
    {
      int kl = tid >> 2, nb = (tid & 3) * 16;
      const float* srcw = W + (size_t)(kb + kl) * N;
#pragma unroll
      for (int j = 0; j < 16; ++j) {
        int n = n0 + nb + j;
        sB[(nb + j) * 40 + kl] = (f16)((n < N) ? srcw[n] : 0.f);
      }
    }
    __syncthreads();
    v16h a = frag_a_lds(sA + (wave * 16 + (lane & 15)) * 40, 0, lane);
#pragma unroll
    for (int nt = 0; nt < 4; ++nt) {
      v16h bf = frag_b_lds(sB + (nt * 16 + (lane & 15)) * 40, 0, lane);
      accs[nt] = wmma_f16(a, bf, accs[nt]);
    }
    __syncthreads();
  }
#pragma unroll
  for (int nt = 0; nt < 4; ++nt) {
    int n = n0 + nt * 16 + (lane & 15);
    if (n < N) {
      float bb = bias[n];
#pragma unroll
      for (int q = 0; q < 8; ++q) {
        int Mg = m0 + wave * 16 + ((lane >> 4) & 1) * 8 + q;
        float v = accs[nt][q] + bb;
        if (relu) v = fmaxf(v, 0.f);
        out[(size_t)Mg * N + n] = (OT)v;
      }
    }
  }
}

// ---------------------------------------------------------------------------
__global__ void concat_kernel(const float* __restrict__ x1,
                              const float* __restrict__ x2,
                              const float* __restrict__ x3,
                              f16* __restrict__ cat, int total) {
  int t = blockIdx.x * blockDim.x + threadIdx.x;
  if (t >= total * 192) return;
  int node = t / 192, c = t % 192;
  float v = (c < 64)  ? x1[(size_t)node * 64 + c]
          : (c < 128) ? x2[(size_t)node * 64 + c - 64]
                      : x3[(size_t)node * 64 + c - 128];
  cat[t] = (f16)v;
}

// ---------------------------------------------------------------------------
__global__ void lsm2_kernel(const float* __restrict__ in, float* __restrict__ out,
                            int total) {
  int t = blockIdx.x * blockDim.x + threadIdx.x;
  if (t >= total) return;
  float a = in[(size_t)t * 2 + 0], b = in[(size_t)t * 2 + 1];
  float m = fmaxf(a, b);
  float lse = m + logf(expf(a - m) + expf(b - m));
  out[(size_t)t * 2 + 0] = a - lse;
  out[(size_t)t * 2 + 1] = b - lse;
}

// ---------------------------------------------------------------------------
__global__ void __launch_bounds__(128)
wrapper_kernel(const float* __restrict__ seg, const float* __restrict__ W1,
               const float* __restrict__ b1, const float* __restrict__ W2,
               const float* __restrict__ b2, float* __restrict__ out) {
  int b = blockIdx.x;
  __shared__ float h[100];
  const float* row = seg + (size_t)b * 2048;
  for (int o = threadIdx.x; o < 100; o += blockDim.x) {
    float acc = b1[o];
    for (int j = 0; j < 2048; ++j) acc += row[j] * W1[(size_t)j * 100 + o];
    h[o] = fmaxf(acc, 0.f);
  }
  __syncthreads();
  if (threadIdx.x == 0) {
    float y0 = b2[0], y1 = b2[1];
    for (int j = 0; j < 100; ++j) {
      y0 += h[j] * W2[j * 2 + 0];
      y1 += h[j] * W2[j * 2 + 1];
    }
    float m = fmaxf(y0, y1);
    float lse = m + logf(expf(y0 - m) + expf(y1 - m));
    out[b * 2 + 0] = y0 - lse;
    out[b * 2 + 1] = y1 - lse;
  }
}

// ---------------------------------------------------------------------------
extern "C" void kernel_launch(void* const* d_in, const int* in_sizes, int n_in,
                              void* d_out, int out_size, void* d_ws, size_t ws_size,
                              hipStream_t stream) {
  (void)in_sizes; (void)n_in; (void)out_size; (void)ws_size;

  const float* pos   = (const float*)d_in[0];
  const float* c1_W1 = (const float*)d_in[3];
  const float* c1_b1 = (const float*)d_in[4];
  const float* c1_s  = (const float*)d_in[5];
  const float* c1_t  = (const float*)d_in[6];
  const float* c1_W2 = (const float*)d_in[7];
  const float* c1_b2 = (const float*)d_in[8];
  const float* c2_W1 = (const float*)d_in[9];
  const float* c2_b1 = (const float*)d_in[10];
  const float* c2_s  = (const float*)d_in[11];
  const float* c2_t  = (const float*)d_in[12];
  const float* c2_W2 = (const float*)d_in[13];
  const float* c2_b2 = (const float*)d_in[14];
  const float* c3_W1 = (const float*)d_in[15];
  const float* c3_b1 = (const float*)d_in[16];
  const float* c3_s  = (const float*)d_in[17];
  const float* c3_t  = (const float*)d_in[18];
  const float* c3_W2 = (const float*)d_in[19];
  const float* c3_b2 = (const float*)d_in[20];
  const float* h_W1  = (const float*)d_in[21];
  const float* h_b1  = (const float*)d_in[22];
  const float* h_W2  = (const float*)d_in[23];
  const float* h_b2  = (const float*)d_in[24];
  const float* h_W3  = (const float*)d_in[25];
  const float* h_b3  = (const float*)d_in[26];
  const float* h_W4  = (const float*)d_in[27];
  const float* h_b4  = (const float*)d_in[28];
  const float* m_W1  = (const float*)d_in[29];
  const float* m_b1  = (const float*)d_in[30];
  const float* m_W2  = (const float*)d_in[31];
  const float* m_b2  = (const float*)d_in[32];

  char* ws = (char*)d_ws;
  size_t off = 0;
  auto alloc = [&](size_t bytes) -> void* {
    void* p = ws + off;
    off += (bytes + 255) & ~(size_t)255;
    return p;
  };

  float* sq     = (float*)alloc((size_t)NODES * 4);
  int*   idx    = (int*)  alloc((size_t)NODES * KNN * 4);
  float* x1     = (float*)alloc((size_t)NODES * 64 * 4);
  float* x2     = (float*)alloc((size_t)NODES * 64 * 4);
  float* x3     = (float*)alloc((size_t)NODES * 64 * 4);
  f16*   cat    = (f16*)  alloc((size_t)NODES * 192 * 2);
  f16*   h1     = (f16*)  alloc((size_t)NODES * 1024 * 2);
  f16*   h2     = (f16*)  alloc((size_t)NODES * 256 * 2);
  f16*   h3     = (f16*)  alloc((size_t)NODES * 128 * 2);
  float* logits = (float*)alloc((size_t)NODES * 2 * 4);
  float* seg    = (float*)alloc((size_t)NODES * 2 * 4);

  const int knn_blocks = BATCH * (NPTS / 16);   // 2048
  auto knn_sh = [](int DP) -> size_t {
    size_t STR = (size_t)DP + 8;
    return (size_t)16 * NPTS * 4 + 16 * STR * 2 + 4 * 16 * STR * 2;
  };
  auto ec_sh = [](int DP) -> size_t {
    size_t ASTR = (size_t)DP + 8;
    return 64 * ASTR * 2 + 64 * 72 * 2 + 8 * 32 * ASTR * 2 + 8 * 32 * 72 * 2;
  };

  // ---- conv1 (d = 7) ----
  sq_kernel<<<(NODES + 255) / 256, 256, 0, stream>>>(pos, sq, NODES, 7);
  knn_kernel<7><<<knn_blocks, 128, knn_sh(32), stream>>>(pos, sq, idx, NPTS);
  edgeconv_kernel<7><<<NODES / 8, 256, ec_sh(32), stream>>>(
      pos, idx, c1_W1, c1_b1, c1_s, c1_t, c1_W2, c1_b2, x1, NPTS);
  // ---- conv2 (d = 64) ----
  sq_kernel<<<(NODES + 255) / 256, 256, 0, stream>>>(x1, sq, NODES, 64);
  knn_kernel<64><<<knn_blocks, 128, knn_sh(64), stream>>>(x1, sq, idx, NPTS);
  edgeconv_kernel<64><<<NODES / 8, 256, ec_sh(128), stream>>>(
      x1, idx, c2_W1, c2_b1, c2_s, c2_t, c2_W2, c2_b2, x2, NPTS);
  // ---- conv3 (d = 64) ----
  sq_kernel<<<(NODES + 255) / 256, 256, 0, stream>>>(x2, sq, NODES, 64);
  knn_kernel<64><<<knn_blocks, 128, knn_sh(64), stream>>>(x2, sq, idx, NPTS);
  edgeconv_kernel<64><<<NODES / 8, 256, ec_sh(128), stream>>>(
      x2, idx, c3_W1, c3_b1, c3_s, c3_t, c3_W2, c3_b2, x3, NPTS);

  // ---- head MLP ----
  concat_kernel<<<((size_t)NODES * 192 + 255) / 256, 256, 0, stream>>>(
      x1, x2, x3, cat, NODES);

  gemm_kernel<f16, f16><<<(NODES / 64) * 16, 128, 0, stream>>>(
      cat, h_W1, h_b1, h1, NODES, 192, 1024, 1, 16);
  gemm_kernel<f16, f16><<<(NODES / 64) * 4, 128, 0, stream>>>(
      h1, h_W2, h_b2, h2, NODES, 1024, 256, 1, 4);
  gemm_kernel<f16, f16><<<(NODES / 64) * 2, 128, 0, stream>>>(
      h2, h_W3, h_b3, h3, NODES, 256, 128, 1, 2);
  gemm_kernel<f16, float><<<(NODES / 64) * 1, 128, 0, stream>>>(
      h3, h_W4, h_b4, logits, NODES, 128, 2, 0, 1);

  lsm2_kernel<<<(NODES + 255) / 256, 256, 0, stream>>>(logits, seg, NODES);

  wrapper_kernel<<<BATCH, 128, 0, stream>>>(seg, m_W1, m_b1, m_W2, m_b2,
                                            (float*)d_out);
}
